// GKATClassifier_27556510171819
// MI455X (gfx1250) — matrix operand order repaired and checked by reading, not compiled, not run on hardware
//
#include <hip/hip_runtime.h>
#include <hip/hip_bf16.h>

typedef __attribute__((ext_vector_type(16))) _Float16 v16h;
typedef __attribute__((ext_vector_type(8)))  float    v8f;
typedef __attribute__((ext_vector_type(4)))  unsigned int v4u;
typedef _Float16 half_t;

#define DEV __device__ __forceinline__

// ---------------------------------------------------------------------------
// WMMA fragment helpers (CDNA5 16x16x32 f16 -> f32)
//
// A fragment (16x32, 16-bit): lane l holds row m = l%16; K values come in
// contiguous pairs at offsets {0,2,4,6,16,18,20,22} + 8*(l/16) -> 8 b32 loads
// B fragment (32x16, 16-bit): lane l holds col n = l%16; K runs of 8 at
// offsets {0,16} + 8*(l/16) (same order as A) -> 2 b128 loads
// ---------------------------------------------------------------------------
union AFrag { unsigned int u[8]; v16h v; };
union BFrag { v4u q[2]; v16h v; };

DEV v16h load_afrag(const half_t* __restrict__ rowp, int kbase, int hs) {
  AFrag a;
  const int base = kbase + hs * 8;
#pragma unroll
  for (int r = 0; r < 8; ++r) {
    int k = base + ((r < 4) ? (2 * r) : (16 + 2 * (r - 4)));
    a.u[r] = *(const unsigned int*)(rowp + k);
  }
  return a.v;
}

DEV v16h load_bfrag(const half_t* __restrict__ colp, int kbase, int hs) {
  BFrag b;
  const half_t* p = colp + kbase + hs * 8;
  b.q[0] = *(const v4u*)p;
  b.q[1] = *(const v4u*)(p + 16);
  return b.v;
}

DEV v8f wmma_f16(v16h a, v16h b, v8f c) {
  return __builtin_amdgcn_wmma_f32_16x16x32_f16(false, a, false, b, (short)0, c,
                                                false, false);
}

// ---------------------------------------------------------------------------
// "NT" GEMM:  C[M,N] = A[M,K] * B[N,K]^T, f16 in, f16 out (f32 accumulate).
// One wave computes a 32(M) x 64(N) tile: 2 A-fragments, 8 accumulators,
// every B fragment feeds two WMMAs. K in steps of 32.
// transStore=1 writes C transposed (C[n*ldc + m]).
// ---------------------------------------------------------------------------
__global__ __launch_bounds__(128) void gemm_nt_f16(
    const half_t* __restrict__ A, const half_t* __restrict__ B,
    half_t* __restrict__ C, int M, int N, int K, int lda, int ldb, int ldc,
    int transStore) {
  const int lane = threadIdx.x & 31;
  const int wave = threadIdx.x >> 5;
  const int nGroups = N >> 6;
  const int gw = blockIdx.x * 4 + wave;
  const int mt = gw / nGroups;
  const int ng = gw % nGroups;
  if (mt * 32 >= M) return;
  const int hs = lane >> 4;
  const int ln = lane & 15;

  const half_t* arow0 = A + (size_t)(mt * 32 + ln) * lda;
  const half_t* arow1 = A + (size_t)(mt * 32 + 16 + ln) * lda;
  v8f acc[2][4] = {};
  for (int kb = 0; kb < K; kb += 32) {
    v16h af0 = load_afrag(arow0, kb, hs);
    v16h af1 = load_afrag(arow1, kb, hs);
#pragma unroll
    for (int nt = 0; nt < 4; ++nt) {
      const half_t* bcol = B + (size_t)(ng * 64 + nt * 16 + ln) * ldb;
      v16h bf = load_bfrag(bcol, kb, hs);
      acc[0][nt] = wmma_f16(af0, bf, acc[0][nt]);
      acc[1][nt] = wmma_f16(af1, bf, acc[1][nt]);
    }
  }
#pragma unroll
  for (int ms = 0; ms < 2; ++ms) {
#pragma unroll
    for (int nt = 0; nt < 4; ++nt) {
      const int n = ng * 64 + nt * 16 + ln;
#pragma unroll
      for (int v = 0; v < 8; ++v) {
        const int m = mt * 32 + ms * 16 + v + 8 * hs;
        half_t val = (half_t)acc[ms][nt][v];
        if (transStore) C[(size_t)n * ldc + m] = val;
        else            C[(size_t)m * ldc + n] = val;
      }
    }
  }
}

// ---------------------------------------------------------------------------
// Fused GKAT attention (per head, 32 Q-rows per wave = two 16-row sub-tiles
// sharing all K / CV B-fragments):
//   num = exp(Q K^T / 8) @ CV ,  den = exp(Q K^T / 8) . rowsumC
//   out = elu(num / (den + 1e-9))
// Max-subtraction cancels exactly in num/den, so it is omitted.
// e tiles go D-layout -> LDS -> A-fragment (stride 36 keeps b32 alignment
// and conflict-free reads).
// ---------------------------------------------------------------------------
__global__ __launch_bounds__(128) void gkat_attn(
    const half_t* __restrict__ Qm, const half_t* __restrict__ Km,
    const half_t* __restrict__ CVT, const float* __restrict__ rowsumC,
    half_t* __restrict__ Hout, int Nnodes, int ldq, int ldh) {
  const int lane = threadIdx.x & 31;
  const int wave = threadIdx.x >> 5;
  const int head = blockIdx.y;
  const int qt = blockIdx.x * 4 + wave;   // 32-row tile index
  const int hs = lane >> 4;
  const int ln = lane & 15;
  const int hoff = head * 64;
  const int q0 = qt * 32;

  __shared__ __attribute__((aligned(16))) half_t etile[4][2][16][36];

  v16h aQ[2][2];
#pragma unroll
  for (int sub = 0; sub < 2; ++sub) {
    const half_t* qrow = Qm + (size_t)(q0 + sub * 16 + ln) * ldq + hoff;
    aQ[sub][0] = load_afrag(qrow, 0, hs);
    aQ[sub][1] = load_afrag(qrow, 32, hs);
  }

  v8f acc[2][4] = {};
  float denp[2][8] = {};

  for (int j0 = 0; j0 < Nnodes; j0 += 32) {
    // ---- stage K fragments once, reuse for both Q sub-tiles ----
    v16h bK[2][2];
    float rs[2];
#pragma unroll
    for (int t = 0; t < 2; ++t) {
      const int key = j0 + t * 16 + ln;
      const half_t* krow = Km + (size_t)key * ldq + hoff;
      bK[t][0] = load_bfrag(krow, 0, hs);
      bK[t][1] = load_bfrag(krow, 32, hs);
      rs[t] = rowsumC[key];
    }

    // ---- S = Q K^T * 0.125 ; e = exp(S) ----
#pragma unroll
    for (int sub = 0; sub < 2; ++sub) {
#pragma unroll
      for (int t = 0; t < 2; ++t) {
        v8f s = {};
        s = wmma_f16(aQ[sub][0], bK[t][0], s);
        s = wmma_f16(aQ[sub][1], bK[t][1], s);
#pragma unroll
        for (int v = 0; v < 8; ++v) {
          float e = __expf(s[v] * 0.125f);
          denp[sub][v] += e * rs[t];
          etile[wave][sub][v + 8 * hs][t * 16 + ln] = (half_t)e;
        }
      }
    }
    asm volatile("" ::: "memory");  // keep LDS store->load order

    // ---- reload e as A fragments ----
    v16h ae[2];
    ae[0] = load_afrag(&etile[wave][0][ln][0], 0, hs);
    ae[1] = load_afrag(&etile[wave][1][ln][0], 0, hs);

    // ---- acc += e @ CV (4 d-tiles, CV fragments shared by both subs) ----
#pragma unroll
    for (int nt = 0; nt < 4; ++nt) {
      const half_t* cvrow = CVT + (size_t)(hoff + nt * 16 + ln) * Nnodes + j0;
      v16h bf = load_bfrag(cvrow, 0, hs);
      acc[0][nt] = wmma_f16(ae[0], bf, acc[0][nt]);
      acc[1][nt] = wmma_f16(ae[1], bf, acc[1][nt]);
    }
    asm volatile("" ::: "memory");
  }

  // reduce denominators over 16 lanes of each half-wave
#pragma unroll
  for (int sub = 0; sub < 2; ++sub) {
#pragma unroll
    for (int v = 0; v < 8; ++v) {
      float d = denp[sub][v];
      d += __shfl_xor(d, 1, 32);
      d += __shfl_xor(d, 2, 32);
      d += __shfl_xor(d, 4, 32);
      d += __shfl_xor(d, 8, 32);
      denp[sub][v] = d + 1e-9f;
    }
  }

  // out = elu(num/den)
#pragma unroll
  for (int sub = 0; sub < 2; ++sub) {
#pragma unroll
    for (int nt = 0; nt < 4; ++nt) {
      const int dcol = hoff + nt * 16 + ln;
#pragma unroll
      for (int v = 0; v < 8; ++v) {
        float r = acc[sub][nt][v] / denp[sub][v];
        r = (r > 0.f) ? r : expm1f(r);
        Hout[(size_t)(q0 + sub * 16 + v + 8 * hs) * ldh + dcol] = (half_t)r;
      }
    }
  }
}

// ---------------------------------------------------------------------------
// Row normalize (mean / std ddof=1) -> f16
// ---------------------------------------------------------------------------
__global__ __launch_bounds__(256) void normalize_rows(
    const float* __restrict__ X, half_t* __restrict__ Xn) {
  __shared__ float red[256];
  const int n = blockIdx.x;
  const float* row = X + (size_t)n * 512;
  float s = 0.f;
  for (int i = threadIdx.x; i < 512; i += 256) s += row[i];
  red[threadIdx.x] = s;
  __syncthreads();
  for (int off = 128; off > 0; off >>= 1) {
    if (threadIdx.x < off) red[threadIdx.x] += red[threadIdx.x + off];
    __syncthreads();
  }
  const float mean = red[0] / 512.f;
  __syncthreads();
  float v = 0.f;
  for (int i = threadIdx.x; i < 512; i += 256) {
    float d = row[i] - mean;
    v += d * d;
  }
  red[threadIdx.x] = v;
  __syncthreads();
  for (int off = 128; off > 0; off >>= 1) {
    if (threadIdx.x < off) red[threadIdx.x] += red[threadIdx.x + off];
    __syncthreads();
  }
  const float inv = 1.f / (sqrtf(red[0] / 511.f) + 1e-9f);
  for (int i = threadIdx.x; i < 512; i += 256)
    Xn[(size_t)n * 512 + i] = (half_t)((row[i] - mean) * inv);
}

// counting -> f16 copy + row sums
__global__ __launch_bounds__(256) void count_prep(
    const float* __restrict__ C, half_t* __restrict__ Cf,
    float* __restrict__ rowsum) {
  __shared__ float red[256];
  const int m = blockIdx.x;
  const float* row = C + (size_t)m * 4096;
  half_t* orow = Cf + (size_t)m * 4096;
  float s = 0.f;
  for (int i = threadIdx.x; i < 4096; i += 256) {
    float x = row[i];
    s += x;
    orow[i] = (half_t)x;
  }
  red[threadIdx.x] = s;
  __syncthreads();
  for (int off = 128; off > 0; off >>= 1) {
    if (threadIdx.x < off) red[threadIdx.x] += red[threadIdx.x + off];
    __syncthreads();
  }
  if (threadIdx.x == 0) rowsum[m] = red[0];
}

__global__ void f32_to_f16(const float* __restrict__ src,
                           half_t* __restrict__ dst, int n) {
  int i = blockIdx.x * blockDim.x + threadIdx.x;
  if (i < n) dst[i] = (half_t)src[i];
}

__global__ __launch_bounds__(256) void classifier_k(
    const half_t* __restrict__ H, const float* __restrict__ Wc,
    const float* __restrict__ bc, float* __restrict__ out) {
  int idx = blockIdx.x * 256 + threadIdx.x;
  if (idx >= 4096 * 10) return;
  const int n = idx / 10, c = idx % 10;
  const half_t* hrow = H + (size_t)n * 64;
  const float* w = Wc + (size_t)c * 64;
  float s = bc[c];
#pragma unroll 8
  for (int d = 0; d < 64; ++d) s += (float)hrow[d] * w[d];
  out[idx] = s;
}

// ---------------------------------------------------------------------------
extern "C" void kernel_launch(void* const* d_in, const int* in_sizes, int n_in,
                              void* d_out, int out_size, void* d_ws,
                              size_t ws_size, hipStream_t stream) {
  (void)in_sizes; (void)n_in; (void)out_size; (void)ws_size;
  const float* feats    = (const float*)d_in[0];
  const float* counting = (const float*)d_in[1];
  const float* Wq1 = (const float*)d_in[2];
  const float* Wk1 = (const float*)d_in[3];
  const float* Wv1 = (const float*)d_in[4];
  const float* Wq2 = (const float*)d_in[5];
  const float* Wk2 = (const float*)d_in[6];
  const float* Wv2 = (const float*)d_in[7];
  const float* Wc  = (const float*)d_in[8];
  const float* bc  = (const float*)d_in[9];
  float* out = (float*)d_out;

  const size_t N = 4096;
  uintptr_t p = (uintptr_t)d_ws;
  auto take = [&](size_t bytes) -> void* {
    p = (p + 255) & ~(uintptr_t)255;
    uintptr_t r = p;
    p += bytes;
    return (void*)r;
  };

  half_t* countF = (half_t*)take(N * N * 2);        // 32 MB, L2-resident
  float*  rowsum = (float*)take(N * 4);
  half_t* Xn     = (half_t*)take(N * 512 * 2);
  half_t* Wq1f   = (half_t*)take(4 * 64 * 512 * 2);
  half_t* Wk1f   = (half_t*)take(4 * 64 * 512 * 2);
  half_t* Wv1f   = (half_t*)take(4 * 64 * 512 * 2);
  half_t* Wq2f   = (half_t*)take(64 * 256 * 2);
  half_t* Wk2f   = (half_t*)take(64 * 256 * 2);
  half_t* Wv2f   = (half_t*)take(64 * 256 * 2);
  half_t* Q1     = (half_t*)take(N * 256 * 2);
  half_t* K1     = (half_t*)take(N * 256 * 2);
  half_t* VT1    = (half_t*)take(256 * N * 2);      // V^T  [256][4096]
  half_t* CVT1   = (half_t*)take(256 * N * 2);      // (counting@V)^T
  half_t* H1     = (half_t*)take(N * 256 * 2);
  half_t* Q2     = (half_t*)take(N * 64 * 2);
  half_t* K2     = (half_t*)take(N * 64 * 2);
  half_t* VT2    = (half_t*)take(64 * N * 2);
  half_t* CVT2   = (half_t*)take(64 * N * 2);
  half_t* H2     = (half_t*)take(N * 64 * 2);

  // prep
  normalize_rows<<<4096, 256, 0, stream>>>(feats, Xn);
  count_prep<<<4096, 256, 0, stream>>>(counting, countF, rowsum);
  f32_to_f16<<<512, 256, 0, stream>>>(Wq1, Wq1f, 4 * 64 * 512);
  f32_to_f16<<<512, 256, 0, stream>>>(Wk1, Wk1f, 4 * 64 * 512);
  f32_to_f16<<<512, 256, 0, stream>>>(Wv1, Wv1f, 4 * 64 * 512);
  f32_to_f16<<<64, 256, 0, stream>>>(Wq2, Wq2f, 64 * 256);
  f32_to_f16<<<64, 256, 0, stream>>>(Wk2, Wk2f, 64 * 256);
  f32_to_f16<<<64, 256, 0, stream>>>(Wv2, Wv2f, 64 * 256);

  // ---- layer 1 ----  (waves = (M/32)*(N/64); 4 waves per block)
  gemm_nt_f16<<<128, 128, 0, stream>>>(Xn, Wq1f, Q1, 4096, 256, 512, 512, 512, 256, 0);
  gemm_nt_f16<<<128, 128, 0, stream>>>(Xn, Wk1f, K1, 4096, 256, 512, 512, 512, 256, 0);
  gemm_nt_f16<<<128, 128, 0, stream>>>(Xn, Wv1f, VT1, 4096, 256, 512, 512, 512, 4096, 1);
  gemm_nt_f16<<<128, 128, 0, stream>>>(countF, VT1, CVT1, 4096, 256, 4096, 4096, 4096, 4096, 1);
  gkat_attn<<<dim3(32, 4), 128, 0, stream>>>(Q1, K1, CVT1, rowsum, H1, 4096, 256, 256);

  // ---- layer 2 ----
  gemm_nt_f16<<<32, 128, 0, stream>>>(H1, Wq2f, Q2, 4096, 64, 256, 256, 256, 64, 0);
  gemm_nt_f16<<<32, 128, 0, stream>>>(H1, Wk2f, K2, 4096, 64, 256, 256, 256, 64, 0);
  gemm_nt_f16<<<32, 128, 0, stream>>>(H1, Wv2f, VT2, 4096, 64, 256, 256, 256, 4096, 1);
  gemm_nt_f16<<<32, 128, 0, stream>>>(countF, VT2, CVT2, 4096, 64, 4096, 4096, 4096, 4096, 1);
  gkat_attn<<<dim3(32, 1), 128, 0, stream>>>(Q2, K2, CVT2, rowsum, H2, 4096, 64, 64);

  // ---- classifier ----
  classifier_k<<<160, 256, 0, stream>>>(H2, Wc, bc, out);
}
